// QM9EDMModel_5935644803736
// MI455X (gfx1250) — matrix operand order repaired and checked by reading, not compiled, not run on hardware
//
#include <hip/hip_runtime.h>
#include <math.h>

// ---------------------------------------------------------------------------
// Problem constants (match reference)
// ---------------------------------------------------------------------------
#define BB 4
#define NN 512
#define DM 256
#define ACH 5
#define PH 128
#define NRBF 16
#define TOK (BB * NN)          // 2048 tokens
#define LN_EPS 1e-5f
#define GAMMA 14.0625f         // (15/4)^2
#define CSTEP (4.0f / 15.0f)

typedef __attribute__((ext_vector_type(16))) __bf16 v16bf;
typedef __attribute__((ext_vector_type(8)))  float  v8f;

union BfVec { unsigned short u[16]; uint4 q[2]; v16bf v; };

__device__ __forceinline__ unsigned short f2bf(float f) {
#if defined(__gfx1250__)
    return __builtin_bit_cast(unsigned short, (__bf16)f);   // native v_cvt
#else
    unsigned int u = __float_as_uint(f);
    unsigned int r = (u + 0x7FFFu + ((u >> 16) & 1u)) >> 16;
    return (unsigned short)r;
#endif
}
// silu via v_rcp_f32 (avoid the IEEE div_scale/div_fixup macro expansion)
__device__ __forceinline__ float fast_silu(float v) {
    return v * __builtin_amdgcn_rcpf(1.0f + __expf(-v));
}

// ---------------------------------------------------------------------------
// Kernel 1: per-token LayerNorm -> bf16, plus raw trunk -> bf16
// ---------------------------------------------------------------------------
__global__ __launch_bounds__(DM) void ln_prep_kernel(
    const float* __restrict__ trunk, const float* __restrict__ g,
    const float* __restrict__ bta,
    unsigned short* __restrict__ xn_bf, unsigned short* __restrict__ trunk_bf)
{
    const int row = blockIdx.x;
    const int tid = threadIdx.x;
    const float x = trunk[row * DM + tid];

    __shared__ float red[DM];
    red[tid] = x;
    __syncthreads();
    for (int s = DM / 2; s > 0; s >>= 1) {
        if (tid < s) red[tid] += red[tid + s];
        __syncthreads();
    }
    const float mu = red[0] * (1.0f / DM);
    __syncthreads();

    const float d = x - mu;
    red[tid] = d * d;
    __syncthreads();
    for (int s = DM / 2; s > 0; s >>= 1) {
        if (tid < s) red[tid] += red[tid + s];
        __syncthreads();
    }
    const float var = red[0] * (1.0f / DM);

    const float xn = d * __builtin_amdgcn_rsqf(var + LN_EPS) * g[tid] + bta[tid];
    xn_bf[row * DM + tid]    = f2bf(xn);
    trunk_bf[row * DM + tid] = f2bf(x);
}

// ---------------------------------------------------------------------------
// Kernel 2: pack K x Nn f32 weights into bf16 WMMA B-fragment order.
// Rows k >= Ksrc are zero (pads w_pr K 16 -> 32). Each lane's fragment is
// 32 contiguous bytes -> two coalesced b128 loads in the GEMMs.
// ---------------------------------------------------------------------------
__global__ void pack_b_kernel(const float* __restrict__ src,
                              unsigned short* __restrict__ dst,
                              int K, int Ksrc, int Nn)
{
    const int idx = blockIdx.x * blockDim.x + threadIdx.x;
    if (idx >= Nn * K) return;
    const int p    = idx & 15;
    const int l    = (idx >> 4) & 31;
    const int frag = idx >> 9;
    const int kt   = K >> 5;
    const int tn   = frag / kt;
    const int kb   = (frag % kt) << 5;
    const int k    = kb + 8 * (l >> 4) + (p & 7) + 16 * (p >> 3);
    const int n    = tn * 16 + (l & 15);
    const float v  = (k < Ksrc) ? src[k * Nn + n] : 0.0f;
    dst[idx] = f2bf(v);
}

// ---------------------------------------------------------------------------
// Kernel 3: WMMA bf16 GEMM, out = act(A @ B + bias). One wave per 16x16 tile.
// Bias rides in the WMMA C-init (broadcast down each column).
// OUT_FRAG=1 stores the accumulator in C-fragment order (2x b128 per lane),
// used for pj so the pair kernel can feed it straight back as a WMMA C tile.
// ---------------------------------------------------------------------------
template <int HAS_BIAS, int ACT_SILU, int OUT_FRAG>
__global__ __launch_bounds__(32) void wmma_gemm_kernel(
    const unsigned short* __restrict__ A, const unsigned short* __restrict__ Bp,
    const float* __restrict__ bias, float* __restrict__ out,
    int M, int Nn, int K)
{
    const int l    = threadIdx.x;
    const int half = l >> 4;
    const int lm   = l & 15;
    const int tm   = blockIdx.x;
    const int tn   = blockIdx.y;
    const int kt   = K >> 5;
    const int n    = tn * 16 + lm;

    v8f acc;
    const float bv = HAS_BIAS ? bias[n] : 0.0f;
#pragma unroll
    for (int r = 0; r < 8; ++r) acc[r] = bv;

    const unsigned short* arow  = A + (size_t)(tm * 16 + lm) * K;
    const unsigned short* bbase = Bp + ((size_t)tn * kt * 32 + l) * 16;

    for (int kb = 0; kb < K; kb += 32) {
        __builtin_prefetch(arow + kb + 64, 0, 0);   // global_prefetch_b8

        BfVec a;
        a.q[0] = *reinterpret_cast<const uint4*>(arow + kb + 8 * half);
        a.q[1] = *reinterpret_cast<const uint4*>(arow + kb + 16 + 8 * half);

        BfVec b;
        const unsigned short* bp = bbase + (size_t)(kb >> 5) * 32 * 16;
        b.q[0] = *reinterpret_cast<const uint4*>(bp);
        b.q[1] = *reinterpret_cast<const uint4*>(bp + 8);

        acc = __builtin_amdgcn_wmma_f32_16x16x32_bf16(
            false, a.v, false, b.v, (short)0, acc, false, false);
    }

    if (ACT_SILU) {
#pragma unroll
        for (int r = 0; r < 8; ++r) acc[r] = fast_silu(acc[r]);
    }

    if (OUT_FRAG) {
        float4 q0 = make_float4(acc[0], acc[1], acc[2], acc[3]);
        float4 q1 = make_float4(acc[4], acc[5], acc[6], acc[7]);
        float4* dst = reinterpret_cast<float4*>(
            out + (((size_t)tm * (Nn >> 4) + tn) * 32 + l) * 8);
        dst[0] = q0;
        dst[1] = q1;
    } else {
#pragma unroll
        for (int r = 0; r < 8; ++r) {
            const int m = tm * 16 + r + 8 * half;
            out[(size_t)m * Nn + n] = acc[r];
        }
    }
}

// ---------------------------------------------------------------------------
// Kernel 4: atom head 2nd layer: out = pad ? 0 : h @ w_a2 + b_a2
// h row staged to LDS via async global->LDS copy (gfx1250 path).
// ---------------------------------------------------------------------------
__global__ __launch_bounds__(DM) void atom2_kernel(
    const float* __restrict__ h, const float* __restrict__ w_a2,
    const float* __restrict__ b_a2, const unsigned char* __restrict__ pad,
    float* __restrict__ out)
{
    const int t = blockIdx.x;
    __shared__ float sh[DM];
#if defined(__gfx1250__)
    {
        const unsigned lds_off = (unsigned)(uintptr_t)&sh[threadIdx.x];
        const unsigned long long ga =
            (unsigned long long)(uintptr_t)(h + (size_t)t * DM + threadIdx.x);
        asm volatile("global_load_async_to_lds_b32 %0, %1, off"
                     :: "v"(lds_off), "v"(ga) : "memory");
        asm volatile("s_wait_asynccnt 0x0" ::: "memory");
    }
#else
    sh[threadIdx.x] = h[(size_t)t * DM + threadIdx.x];
#endif
    __syncthreads();
    if (threadIdx.x < ACH) {
        const int a = threadIdx.x;
        float s = b_a2[a];
        for (int c = 0; c < DM; ++c) s += sh[c] * w_a2[c * ACH + a];
        out[t * ACH + a] = pad[t] ? 0.0f : s;
    }
}

// ---------------------------------------------------------------------------
// Kernel 5: fused pair head + coord accumulation.
// Block = (b, 16 i's); 8 waves stripe the 32 j-tiles. Per (i, j-tile):
// build 16(j) x 32(K) rbf A-fragment in regs, 8 WMMAs against resident
// packed w_pr fragments with the pj tile riding in as the WMMA C operand
// (D = rbf @ w_pr + pj), then fuse pi+bias, SiLU, dot(w_p2) via half-wave
// shfl reduction, mask, ds_add_f32 into per-i LDS accumulators.
// ---------------------------------------------------------------------------
__global__ __launch_bounds__(256, 1) void pair_kernel(
    const float* __restrict__ coords, const unsigned char* __restrict__ pad,
    const float* __restrict__ pi_g, const float* __restrict__ pj_f,
    const unsigned short* __restrict__ wprp, const float* __restrict__ b_p1,
    const float* __restrict__ w_p2, const float* __restrict__ b_p2,
    float* __restrict__ coord_acc)
{
    const int b     = blockIdx.x >> 5;
    const int itile = blockIdx.x & 31;
    const int tid   = threadIdx.x;
    const int wave  = tid >> 5;
    const int l     = tid & 31;
    const int half  = l >> 4;
    const int lm    = l & 15;

    __shared__ float ci[16][3];
    __shared__ float cacc[16][3];
    __shared__ unsigned char vi[16];

    if (tid < 48) {
        const int i = tid / 3, c = tid % 3;
        ci[i][c]   = coords[(b * NN + itile * 16 + i) * 3 + c];
        cacc[i][c] = 0.0f;
    }
    if (tid < 16) vi[tid] = pad[b * NN + itile * 16 + tid] ? 0 : 1;
    __syncthreads();

    // Resident packed w_pr B fragments (K=32 covers all), 8 n-tiles.
    v16bf bfr[8];
#pragma unroll
    for (int t = 0; t < 8; ++t) {
        BfVec tmp;
        const unsigned short* bp = wprp + ((size_t)t * 32 + l) * 16;
        tmp.q[0] = *reinterpret_cast<const uint4*>(bp);
        tmp.q[1] = *reinterpret_cast<const uint4*>(bp + 8);
        bfr[t] = tmp.v;
    }
    // per-lane channel constants (ch = t*16 + lm)
    float bp1v[8], wpv[8];
#pragma unroll
    for (int t = 0; t < 8; ++t) {
        bp1v[t] = b_p1[t * 16 + lm];
        wpv[t]  = w_p2[t * 16 + lm];
    }
    const float bp2 = b_p2[0];

    for (int jt = wave; jt < NN / 16; jt += 8) {
        const int jbase = jt * 16;
        // lane's j-row for the A fragment (rows m = lane&15 in both halves)
        const float cj0 = coords[(b * NN + jbase + lm) * 3 + 0];
        const float cj1 = coords[(b * NN + jbase + lm) * 3 + 1];
        const float cj2 = coords[(b * NN + jbase + lm) * 3 + 2];

        // pj tile in WMMA C-fragment order: two b128 loads per n-tile
        v8f pjt[8];
#pragma unroll
        for (int t = 0; t < 8; ++t) {
            const float4* q = reinterpret_cast<const float4*>(
                pj_f + (((size_t)(b * 32 + jt) * 8 + t) * 32 + l) * 8);
            const float4 q0 = q[0], q1 = q[1];
            pjt[t][0] = q0.x; pjt[t][1] = q0.y; pjt[t][2] = q0.z; pjt[t][3] = q0.w;
            pjt[t][4] = q1.x; pjt[t][5] = q1.y; pjt[t][6] = q1.z; pjt[t][7] = q1.w;
        }

#pragma unroll 1
        for (int il = 0; il < 16; ++il) {
            const int ig = itile * 16 + il;
            const float dx = ci[il][0] - cj0;
            const float dy = ci[il][1] - cj1;
            const float dz = ci[il][2] - cj2;
            const float sq = dx * dx + dy * dy + dz * dz;
            const float dist = __builtin_amdgcn_sqrtf(fmaxf(sq, 1e-12f));

            // A fragment: p<8 -> rbf k = p + 8*half; p>=8 -> zero padding
            BfVec a;
#pragma unroll
            for (int p = 0; p < 8; ++p) {
                const float d = dist - CSTEP * (float)(p + 8 * half);
                a.u[p]     = f2bf(__expf(-GAMMA * d * d));
                a.u[8 + p] = 0;
            }

            float sumr[8];
#pragma unroll
            for (int r = 0; r < 8; ++r) sumr[r] = 0.0f;

            const float* pig = pi_g + (size_t)(b * NN + ig) * PH;
#pragma unroll
            for (int t = 0; t < 8; ++t) {
                // D = rbf @ w_pr + pj   (pj tile as the C operand)
                v8f acc = __builtin_amdgcn_wmma_f32_16x16x32_bf16(
                    false, a.v, false, bfr[t], (short)0, pjt[t], false, false);
                const float pich = pig[t * 16 + lm] + bp1v[t];
#pragma unroll
                for (int r = 0; r < 8; ++r)
                    sumr[r] += fast_silu(acc[r] + pich) * wpv[t];
            }
            // reduce the 128 channels across the 16 lanes of each half
#pragma unroll
            for (int off = 1; off < 16; off <<= 1) {
#pragma unroll
                for (int r = 0; r < 8; ++r)
                    sumr[r] += __shfl_xor(sumr[r], off, 32);
            }

            // one lane per row accumulates pw * delta into LDS (ds_add_f32)
#pragma unroll
            for (int r = 0; r < 8; ++r) {
                if (lm == r) {
                    const int jg = jbase + r + 8 * half;
                    const bool pv = vi[il] && !pad[b * NN + jg] && (ig != jg);
                    const float pw = pv ? (sumr[r] + bp2) : 0.0f;
                    const float jx = coords[(b * NN + jg) * 3 + 0];
                    const float jy = coords[(b * NN + jg) * 3 + 1];
                    const float jz = coords[(b * NN + jg) * 3 + 2];
                    atomicAdd(&cacc[il][0], pw * (ci[il][0] - jx));
                    atomicAdd(&cacc[il][1], pw * (ci[il][1] - jy));
                    atomicAdd(&cacc[il][2], pw * (ci[il][2] - jz));
                }
            }
        }
    }
    __syncthreads();
    if (tid < 48) {
        const int i = tid / 3, c = tid % 3;
        coord_acc[(b * NN + itile * 16 + i) * 3 + c] = cacc[i][c];
    }
}

// ---------------------------------------------------------------------------
// Kernel 6: finalize coords — /denom, mask, mean-center, mask.
// ---------------------------------------------------------------------------
__global__ __launch_bounds__(NN) void finalize_kernel(
    const float* __restrict__ coord_acc, const unsigned char* __restrict__ pad,
    float* __restrict__ out)
{
    const int b = blockIdx.x;
    const int i = threadIdx.x;
    const bool valid = !pad[b * NN + i];
    const float vf = valid ? 1.0f : 0.0f;

    __shared__ float sbuf[NN];
    auto block_sum = [&](float v) -> float {
        sbuf[i] = v;
        __syncthreads();
        for (int s = NN / 2; s > 0; s >>= 1) {
            if (i < s) sbuf[i] += sbuf[i + s];
            __syncthreads();
        }
        const float r = sbuf[0];
        __syncthreads();
        return r;
    };

    const float denom = fmaxf(block_sum(vf), 1.0f);
    const float rden  = __builtin_amdgcn_rcpf(denom);
    float cd[3];
#pragma unroll
    for (int c = 0; c < 3; ++c) {
        const float raw = coord_acc[(b * NN + i) * 3 + c];
        cd[c] = valid ? raw * rden : 0.0f;
    }
    float mean[3];
#pragma unroll
    for (int c = 0; c < 3; ++c) mean[c] = block_sum(cd[c] * vf) * rden;

    const int base = BB * NN * ACH;
#pragma unroll
    for (int c = 0; c < 3; ++c)
        out[base + (b * NN + i) * 3 + c] = valid ? (cd[c] - mean[c]) : 0.0f;
}

// ---------------------------------------------------------------------------
// Host orchestration
// ---------------------------------------------------------------------------
extern "C" void kernel_launch(void* const* d_in, const int* in_sizes, int n_in,
                              void* d_out, int out_size, void* d_ws, size_t ws_size,
                              hipStream_t stream)
{
    const float*         trunk  = (const float*)d_in[0];
    const float*         coords = (const float*)d_in[1];
    const unsigned char* pad    = (const unsigned char*)d_in[2];   // bool mask
    const float*         ln_g   = (const float*)d_in[3];
    const float*         ln_b   = (const float*)d_in[4];
    const float*         w_a1   = (const float*)d_in[5];
    const float*         b_a1   = (const float*)d_in[6];
    const float*         w_a2   = (const float*)d_in[7];
    const float*         b_a2   = (const float*)d_in[8];
    const float*         w_pi   = (const float*)d_in[9];
    const float*         w_pj   = (const float*)d_in[10];
    const float*         w_pr   = (const float*)d_in[11];
    const float*         b_p1   = (const float*)d_in[12];
    const float*         w_p2   = (const float*)d_in[13];
    const float*         b_p2   = (const float*)d_in[14];
    float* out = (float*)d_out;

    // workspace carve-up (all chunk sizes multiples of 256B)
    char* w = (char*)d_ws;
    unsigned short* trunk_bf = (unsigned short*)w; w += (size_t)TOK * DM * 2;
    unsigned short* xn_bf    = (unsigned short*)w; w += (size_t)TOK * DM * 2;
    unsigned short* wa1p     = (unsigned short*)w; w += (size_t)DM * DM * 2;
    unsigned short* wpip     = (unsigned short*)w; w += (size_t)DM * PH * 2;
    unsigned short* wpjp     = (unsigned short*)w; w += (size_t)DM * PH * 2;
    unsigned short* wprp     = (unsigned short*)w; w += (size_t)32 * PH * 2;
    float* h_buf     = (float*)w; w += (size_t)TOK * DM * 4;
    float* pi_buf    = (float*)w; w += (size_t)TOK * PH * 4;
    float* pj_frag   = (float*)w; w += (size_t)TOK * PH * 4;   // fragment order
    float* coord_acc = (float*)w; w += (size_t)BB * NN * 3 * 4;

    // 1) LayerNorm + bf16 conversions of activations
    ln_prep_kernel<<<TOK, DM, 0, stream>>>(trunk, ln_g, ln_b, xn_bf, trunk_bf);

    // 2) pack weights into WMMA B-fragment order (bf16)
    pack_b_kernel<<<(DM * DM + 255) / 256, 256, 0, stream>>>(w_a1, wa1p, DM, DM, DM);
    pack_b_kernel<<<(DM * PH + 255) / 256, 256, 0, stream>>>(w_pi, wpip, DM, DM, PH);
    pack_b_kernel<<<(DM * PH + 255) / 256, 256, 0, stream>>>(w_pj, wpjp, DM, DM, PH);
    pack_b_kernel<<<(32 * PH + 255) / 256, 256, 0, stream>>>(w_pr, wprp, 32, NRBF, PH);

    // 3) h = silu(xn @ w_a1 + b_a1)        [2048 x 256 x 256]
    wmma_gemm_kernel<1, 1, 0><<<dim3(TOK / 16, DM / 16), 32, 0, stream>>>(
        xn_bf, wa1p, b_a1, h_buf, TOK, DM, DM);

    // 4) pi = trunk @ w_pi (row-major) ; pj = trunk @ w_pj (fragment order)
    wmma_gemm_kernel<0, 0, 0><<<dim3(TOK / 16, PH / 16), 32, 0, stream>>>(
        trunk_bf, wpip, nullptr, pi_buf, TOK, PH, DM);
    wmma_gemm_kernel<0, 0, 1><<<dim3(TOK / 16, PH / 16), 32, 0, stream>>>(
        trunk_bf, wpjp, nullptr, pj_frag, TOK, PH, DM);

    // 5) atom_delta -> out[0 : TOK*5]
    atom2_kernel<<<TOK, DM, 0, stream>>>(h_buf, w_a2, b_a2, pad, out);

    // 6) fused pair head + coord accumulation
    pair_kernel<<<BB * (NN / 16), 256, 0, stream>>>(
        coords, pad, pi_buf, pj_frag, wprp, b_p1, w_p2, b_p2, coord_acc);

    // 7) finalize coord_delta -> out[TOK*5 : TOK*5 + TOK*3]
    finalize_kernel<<<BB, NN, 0, stream>>>(coord_acc, pad, out);
}
